// Ensemble_25941602468195
// MI455X (gfx1250) — matrix operand order repaired and checked by reading, not compiled, not run on hardware
//
#include <hip/hip_runtime.h>

typedef _Float16 h16;
typedef __attribute__((ext_vector_type(4)))  _Float16 v4h;
typedef __attribute__((ext_vector_type(8)))  _Float16 v8h;
typedef __attribute__((ext_vector_type(16))) _Float16 v16h;
typedef __attribute__((ext_vector_type(4)))  float    v4f;
typedef __attribute__((ext_vector_type(8)))  float    v8f;

#define IN_SIZE   64
#define AC_BASE   48   // IN_SIZE - AC_SIZE, columns >= 48 get clipped
#define OUT_SIZE  48
#define DEPTH     128
#define NEXPERT   8
#define BATCH     65536

#define NWAVES          8
#define TPB             (NWAVES * 32)      // 256 threads
#define TILES_PER_WAVE  4
#define ROWS_PER_WG     (NWAVES * TILES_PER_WAVE * 16)   // 512 rows

#define SHUF16(lo, hi) __builtin_shufflevector(lo, hi, 0,1,2,3,4,5,6,7,8,9,10,11,12,13,14,15)
#define MEMFENCE() asm volatile("" ::: "memory")

// LDS temp for natural-layout f16 weights, padded stride (N+2) so the
// column-gather in the swizzle phase walks consecutive banks.
#define TEMP_HALFS (DEPTH * (DEPTH + 2))   // 128*130 = 16640 halfs = 33280 B

// Two-phase weight staging:
//  A) coalesced global f32 read -> f16 -> natural [k][n] layout in padded temp
//  B) LDS->LDS swizzle into WMMA B-operand layout (16-bit B, 32x16 tiles):
//     for tile (ktile,ntile), lane L holds the 16 halfs
//     W[ktile*32 + (j>=8 ? 16:0) + (L>=16 ? 8:0) + (j&7)][ntile*16 + (L&15)], j = 0..15.
__device__ __forceinline__ void stage_matrix(const float* __restrict__ W, h16* dst,
                                             int K, int N, int NT, int tid, h16* temp) {
    const int NP = N + 2;
    const int total = K * N;
    for (int idx = tid; idx < total; idx += TPB) {
        int k = idx / N, n = idx - k * N;
        temp[k * NP + n] = (h16)W[idx];
    }
    __syncthreads();
    for (int p = tid; p < total; p += TPB) {
        int j     = p & 15;
        int lane  = (p >> 4) & 31;
        int tile  = p >> 9;
        int ntile = tile % NT;
        int ktile = tile / NT;
        int k = ktile * 32 + ((j >> 3) << 4) + ((lane >> 4) << 3) + (j & 7);
        int n = ntile * 16 + (lane & 15);
        dst[p] = temp[k * NP + n];
    }
    __syncthreads();
}

__global__ __launch_bounds__(TPB) void ensemble_mlp_wmma(
    const float* __restrict__ in,
    const float* __restrict__ W1, const float* __restrict__ b1,
    const float* __restrict__ W2, const float* __restrict__ b2,
    const float* __restrict__ W3, const float* __restrict__ b3,
    float* __restrict__ out)
{
    // Weights in B-operand layout: one v16h (32 B) per lane per (ktile,ntile)
    __shared__ v16h sW1[2 * 8 * 32];   // 64x128  -> 16 KB
    __shared__ v16h sW2[4 * 8 * 32];   // 128x128 -> 32 KB
    __shared__ v16h sW3[4 * 3 * 32];   // 128x48  -> 12 KB
    __shared__ float sB1[DEPTH], sB2[DEPTH], sB3[OUT_SIZE];
    // Staging temp overlaid with the per-wave activation scratch (not live
    // simultaneously): 33280 B.
    __shared__ __align__(16) union {
        h16 temp[TEMP_HALFS];
        h16 hb[NWAVES][16][DEPTH];
    } u;

    const int tid  = threadIdx.x;
    const int lane = tid & 31;
    const int wv   = tid >> 5;
    const int e    = blockIdx.y;
    const int rowbase_wg = blockIdx.x * ROWS_PER_WG;

    // ---------------- stage weights + biases into LDS ----------------
    stage_matrix(W1 + (size_t)e * IN_SIZE * DEPTH,    (h16*)sW1, IN_SIZE, DEPTH,    8, tid, u.temp);
    stage_matrix(W2 + (size_t)e * DEPTH * DEPTH,      (h16*)sW2, DEPTH,   DEPTH,    8, tid, u.temp);
    stage_matrix(W3 + (size_t)e * DEPTH * OUT_SIZE,   (h16*)sW3, DEPTH,   OUT_SIZE, 3, tid, u.temp);
    for (int i = tid; i < DEPTH; i += TPB) { sB1[i] = b1[e * DEPTH + i]; sB2[i] = b2[e * DEPTH + i]; }
    for (int i = tid; i < OUT_SIZE; i += TPB) sB3[i] = b3[e * OUT_SIZE + i];
    __syncthreads();

    // A-operand addressing (16-bit A 16x32): lane<16 holds row=lane, K off 0;
    // lane>=16 holds row=lane-16, K off 8 (each half-chunk of 8 contiguous K).
    const int arow = lane & 15;
    const int koff = (lane >> 4) * 8;
    const int ncol = lane & 15;          // C/D & B column within tile
    const int mrow = (lane >> 4) * 8;    // C/D row base (VGPR r -> M = mrow + r)
    const int rowq  = lane >> 1;         // input load: 2 lanes per row
    const int cbase = (lane & 1) * 32;

    for (int t = 0; t < TILES_PER_WAVE; ++t) {
        const int tilebase = rowbase_wg + (wv * TILES_PER_WAVE + t) * 16;

        // ---- load 16x64 fp32 inputs, clip last 16 cols, convert to f16 ----
        {
            const float* src = in + ((size_t)e * BATCH + tilebase + rowq) * IN_SIZE + cbase;
            h16* dstrow = &u.hb[wv][rowq][cbase];
#pragma unroll
            for (int i = 0; i < 8; ++i) {
                v4f v = *(const v4f*)(src + 4 * i);
                if (cbase + 4 * i >= AC_BASE) {
#pragma unroll
                    for (int c = 0; c < 4; ++c)
                        v[c] = fminf(fmaxf(v[c], -1.0f), 1.0f);
                }
                v4h h;
#pragma unroll
                for (int c = 0; c < 4; ++c) h[c] = (h16)v[c];
                *(v4h*)(dstrow + 4 * i) = h;
            }
        }
        MEMFENCE();

        // ---- layer 1: (16x64) @ (64x128) + b1, ReLU ----
        v16h a0, a1;
        {
            const h16* r = u.hb[wv][arow];
            v8h lo0 = *(const v8h*)(r + koff);
            v8h hi0 = *(const v8h*)(r + 16 + koff);
            v8h lo1 = *(const v8h*)(r + 32 + koff);
            v8h hi1 = *(const v8h*)(r + 48 + koff);
            a0 = SHUF16(lo0, hi0);
            a1 = SHUF16(lo1, hi1);
        }
        MEMFENCE();
#pragma unroll
        for (int nt = 0; nt < 8; ++nt) {
            v8f acc = {};
            acc = __builtin_amdgcn_wmma_f32_16x16x32_f16(false, a0, false, sW1[(0 * 8 + nt) * 32 + lane], (short)0, acc, false, false);
            acc = __builtin_amdgcn_wmma_f32_16x16x32_f16(false, a1, false, sW1[(1 * 8 + nt) * 32 + lane], (short)0, acc, false, false);
            const int n = nt * 16 + ncol;
            const float bias = sB1[n];
#pragma unroll
            for (int r8 = 0; r8 < 8; ++r8) {
                float v = acc[r8] + bias;
                u.hb[wv][mrow + r8][n] = (h16)(v > 0.0f ? v : 0.0f);
            }
        }
        MEMFENCE();

        // ---- layer 2: (16x128) @ (128x128) + b2, ReLU ----
        v16h A2[4];
#pragma unroll
        for (int c = 0; c < 4; ++c) {
            const h16* r = u.hb[wv][arow];
            v8h lo = *(const v8h*)(r + c * 32 + koff);
            v8h hi = *(const v8h*)(r + c * 32 + 16 + koff);
            A2[c] = SHUF16(lo, hi);
        }
        MEMFENCE();
#pragma unroll
        for (int nt = 0; nt < 8; ++nt) {
            v8f acc = {};
#pragma unroll
            for (int c = 0; c < 4; ++c)
                acc = __builtin_amdgcn_wmma_f32_16x16x32_f16(false, A2[c], false, sW2[(c * 8 + nt) * 32 + lane], (short)0, acc, false, false);
            const int n = nt * 16 + ncol;
            const float bias = sB2[n];
#pragma unroll
            for (int r8 = 0; r8 < 8; ++r8) {
                float v = acc[r8] + bias;
                u.hb[wv][mrow + r8][n] = (h16)(v > 0.0f ? v : 0.0f);
            }
        }
        MEMFENCE();

        // ---- layer 3: (16x128) @ (128x48) + b3 -> global ----
        v16h A3[4];
#pragma unroll
        for (int c = 0; c < 4; ++c) {
            const h16* r = u.hb[wv][arow];
            v8h lo = *(const v8h*)(r + c * 32 + koff);
            v8h hi = *(const v8h*)(r + c * 32 + 16 + koff);
            A3[c] = SHUF16(lo, hi);
        }
        MEMFENCE();
#pragma unroll
        for (int nt = 0; nt < 3; ++nt) {
            v8f acc = {};
#pragma unroll
            for (int c = 0; c < 4; ++c)
                acc = __builtin_amdgcn_wmma_f32_16x16x32_f16(false, A3[c], false, sW3[(c * 3 + nt) * 32 + lane], (short)0, acc, false, false);
            const int n = nt * 16 + ncol;
            const float bias = sB3[n];
#pragma unroll
            for (int r8 = 0; r8 < 8; ++r8) {
                out[((size_t)e * BATCH + tilebase + mrow + r8) * OUT_SIZE + n] = acc[r8] + bias;
            }
        }
    }
}

extern "C" void kernel_launch(void* const* d_in, const int* in_sizes, int n_in,
                              void* d_out, int out_size, void* d_ws, size_t ws_size,
                              hipStream_t stream) {
    const float* in = (const float*)d_in[0];
    const float* W1 = (const float*)d_in[1];
    const float* b1 = (const float*)d_in[2];
    const float* W2 = (const float*)d_in[3];
    const float* b2 = (const float*)d_in[4];
    const float* W3 = (const float*)d_in[5];
    const float* b3 = (const float*)d_in[6];
    float* out = (float*)d_out;

    dim3 grid(BATCH / ROWS_PER_WG, NEXPERT);
    ensemble_mlp_wmma<<<grid, TPB, 0, stream>>>(in, W1, b1, W2, b2, W3, b3, out);
}
